// ImprovedGAFA_PPF_EdgeConv_5274219840020
// MI455X (gfx1250) — matrix operand order repaired and checked by reading, not compiled, not run on hardware
//
#include <hip/hip_runtime.h>
#include <math.h>

// Problem constants (from reference setup_inputs)
#define B_    8
#define KPT_  512
#define NPTS_ 2048
#define C_    128
#define R1_   (B_*KPT_)      // 4096 keypoint rows
#define R8_   (R1_*8)        // 32768 (kpt x k_graph)
#define R16_  (R1_*16)       // 65536 (kpt x 16 neighbors)
#define EPS_BN 1e-5f

typedef unsigned short u16;
typedef __attribute__((ext_vector_type(16))) __bf16  v16bf;
typedef __attribute__((ext_vector_type(8)))  float   v8f;
typedef __attribute__((ext_vector_type(8)))  u16     vu8;
typedef __attribute__((ext_vector_type(16))) u16     vu16;

// f32 -> bf16 (round-to-nearest-even), bit-level so no reliance on fp conv codegen
static __device__ inline u16 f2bf(float f) {
  union { float f; unsigned u; } v; v.f = f;
  return (u16)((v.u + 0x7FFFu + ((v.u >> 16) & 1u)) >> 16);
}
static __device__ inline float bf2f(u16 s) {
  union { unsigned u; float f; } v; v.u = ((unsigned)s) << 16;
  return v.f;
}

// ---------------------------------------------------------------------------
// WMMA GEMM:  C[M,N] = A[M,K] @ W[N,K]^T + bias   (x @ w.T + b)
// A, W are bf16-resident in memory (packed b128 loads). f32 accumulate.
// One wave computes a 16x64 tile: one A fragment feeds 4 WMMAs per k-step.
// Layouts used here (A/B in contiguous k order, not AMD's table order):
//   A 16x32 : lane l (m=l&15, h=l>>4) holds K = h*16+0..15
//   B 32x16 : lane l (n=l&15, h=l>>4) holds K = h*16+0..15
//   C/D     : lane l, vgpr v -> (m = v + 8*(l>>4), n = l&15)
// Requires: M%16==0, N%64==0, K%32==0 (true for every GEMM in this net).
// ---------------------------------------------------------------------------
__global__ void k_gemm_wmma_bf16(const u16* __restrict__ A,
                                 const u16* __restrict__ W,
                                 const float* __restrict__ bias,
                                 void* __restrict__ Cptr,
                                 int M, int N, int K, int ldC, int relu, int outBf16) {
  const int wave = threadIdx.x >> 5;
  const int lane = threadIdx.x & 31;
  const int tilesN = N >> 6;
  const int tile = blockIdx.x * (blockDim.x >> 5) + wave;
  const int totalTiles = (M >> 4) * tilesN;
  if (tile >= totalTiles) return;                 // wave-uniform exit
  const int tm = tile / tilesN;
  const int tn = tile - tm * tilesN;
  const int m0 = tm << 4, n0 = tn << 6;
  const int half = lane >> 4;                     // 0 or 1
  const int r    = lane & 15;

  v8f acc[4] = {{}, {}, {}, {}};
  const u16* Arow = A + (size_t)(m0 + r) * K;
  const u16* Wrow[4];
#pragma unroll
  for (int i = 0; i < 4; ++i) Wrow[i] = W + (size_t)(n0 + i * 16 + r) * K;

  for (int k0 = 0; k0 < K; k0 += 32) {
    vu8 alo = *(const vu8*)(Arow + k0 + half * 16);
    vu8 ahi = *(const vu8*)(Arow + k0 + half * 16 + 8);
    v16bf a = __builtin_bit_cast(
        v16bf, __builtin_shufflevector(alo, ahi, 0, 1, 2, 3, 4, 5, 6, 7,
                                       8, 9, 10, 11, 12, 13, 14, 15));
    const int wo = k0 + half * 16;
#pragma unroll
    for (int i = 0; i < 4; ++i) {
      v16bf b = __builtin_bit_cast(v16bf, *(const vu16*)(Wrow[i] + wo));
      acc[i] = __builtin_amdgcn_wmma_f32_16x16x32_bf16(
          /*neg_a=*/false, a, /*neg_b=*/false, b,
          /*c_mod=*/(short)0, acc[i], /*reuse_a=*/false, /*reuse_b=*/false);
    }
  }

#pragma unroll
  for (int i = 0; i < 4; ++i) {
    const int n = n0 + i * 16 + r;
    const float bv = bias ? bias[n] : 0.0f;
#pragma unroll
    for (int v = 0; v < 8; ++v) {
      const int m = m0 + v + 8 * half;
      float val = acc[i][v] + bv;
      if (relu) val = fmaxf(val, 0.0f);
      if (outBf16) ((u16*)Cptr)[(size_t)m * ldC + n] = f2bf(val);
      else         ((float*)Cptr)[(size_t)m * ldC + n] = val;
    }
  }
}

// f32 -> bf16 bulk convert (weights, kpt features)
__global__ void k_cvt_bf16(const float* __restrict__ X, u16* __restrict__ Y, long long n) {
  long long t = (long long)blockIdx.x * blockDim.x + threadIdx.x;
  if (t < n) Y[t] = f2bf(X[t]);
}

// ---------------------------------------------------------------------------
// Analytic symmetric 3x3 eigen (trig method); ev[0] <= ev[1] <= ev[2]
// ---------------------------------------------------------------------------
static __device__ void eig3_sym(float xx, float xy, float xz,
                                float yy, float yz, float zz, float ev[3]) {
  float p1 = xy * xy + xz * xz + yz * yz;
  if (p1 < 1e-24f) {
    float lo = fminf(xx, fminf(yy, zz)), hi = fmaxf(xx, fmaxf(yy, zz));
    ev[0] = lo; ev[2] = hi; ev[1] = xx + yy + zz - lo - hi; return;
  }
  float q = (xx + yy + zz) * (1.0f / 3.0f);
  float p2 = (xx-q)*(xx-q) + (yy-q)*(yy-q) + (zz-q)*(zz-q) + 2.0f * p1;
  float p = sqrtf(p2 * (1.0f / 6.0f));
  float inv = 1.0f / p;
  float bxx=(xx-q)*inv, bxy=xy*inv, bxz=xz*inv, byy=(yy-q)*inv, byz=yz*inv, bzz=(zz-q)*inv;
  float detB = bxx*(byy*bzz - byz*byz) - bxy*(bxy*bzz - byz*bxz) + bxz*(bxy*byz - byy*bxz);
  float rr = fminf(fmaxf(detB * 0.5f, -1.0f), 1.0f);
  float phi = acosf(rr) * (1.0f / 3.0f);
  float e2 = q + 2.0f * p * cosf(phi);
  float e0 = q + 2.0f * p * cosf(phi + 2.0943951023931953f);
  ev[0] = e0; ev[2] = e2; ev[1] = 3.0f * q - e0 - e2;
}

static __device__ void evec_smallest(float xx, float xy, float xz,
                                     float yy, float yz, float zz,
                                     float lam, float out[3]) {
  float r0x = xx - lam, r0y = xy,       r0z = xz;
  float r1x = xy,       r1y = yy - lam, r1z = yz;
  float r2x = xz,       r2y = yz,       r2z = zz - lam;
  float c0x = r0y*r1z - r0z*r1y, c0y = r0z*r1x - r0x*r1z, c0z = r0x*r1y - r0y*r1x;
  float c1x = r0y*r2z - r0z*r2y, c1y = r0z*r2x - r0x*r2z, c1z = r0x*r2y - r0y*r2x;
  float c2x = r1y*r2z - r1z*r2y, c2y = r1z*r2x - r1x*r2z, c2z = r1x*r2y - r1y*r2x;
  float n0 = c0x*c0x + c0y*c0y + c0z*c0z;
  float n1 = c1x*c1x + c1y*c1y + c1z*c1z;
  float n2 = c2x*c2x + c2y*c2y + c2z*c2z;
  float vx = c0x, vy = c0y, vz = c0z, nb = n0;
  if (n1 > nb) { vx = c1x; vy = c1y; vz = c1z; nb = n1; }
  if (n2 > nb) { vx = c2x; vy = c2y; vz = c2z; nb = n2; }
  if (nb < 1e-30f) { vx = 0.0f; vy = 0.0f; vz = 1.0f; nb = 1.0f; }
  float s = rsqrtf(nb);
  out[0] = vx * s; out[1] = vy * s; out[2] = vz * s;
}

// ---------------------------------------------------------------------------
// Normal estimation: top-30 NN (excluding self) -> covariance -> min-evec
// ---------------------------------------------------------------------------
__global__ void k_normals(const float* __restrict__ pts, float* __restrict__ normals) {
  int t = blockIdx.x * blockDim.x + threadIdx.x;
  if (t >= B_ * NPTS_) return;
  int b = t / NPTS_, i = t - b * NPTS_;
  const float* P = pts + (size_t)b * NPTS_ * 3;
  float px = P[i*3], py = P[i*3+1], pz = P[i*3+2];
  const int K = 30;
  float bd[K]; int bi[K];
  for (int q = 0; q < K; ++q) { bd[q] = 3.4e38f; bi[q] = 0; }
  for (int j = 0; j < NPTS_; ++j) {
    if (j == i) continue;
    float dx = P[j*3]-px, dy = P[j*3+1]-py, dz = P[j*3+2]-pz;
    float d = dx*dx + dy*dy + dz*dz;
    if (d < bd[K-1]) {
      int q = K - 1;
      while (q > 0 && bd[q-1] > d) { bd[q] = bd[q-1]; bi[q] = bi[q-1]; --q; }
      bd[q] = d; bi[q] = j;
    }
  }
  float mx = 0, my = 0, mz = 0;
  for (int q = 0; q < K; ++q) { int id = bi[q]; mx += P[id*3]; my += P[id*3+1]; mz += P[id*3+2]; }
  mx *= (1.0f/K); my *= (1.0f/K); mz *= (1.0f/K);
  float cxx=0,cxy=0,cxz=0,cyy=0,cyz=0,czz=0;
  for (int q = 0; q < K; ++q) {
    int id = bi[q];
    float dx = P[id*3]-mx, dy = P[id*3+1]-my, dz = P[id*3+2]-mz;
    cxx += dx*dx; cxy += dx*dy; cxz += dx*dz; cyy += dy*dy; cyz += dy*dz; czz += dz*dz;
  }
  float s = 1.0f / K;
  cxx*=s; cxy*=s; cxz*=s; cyy*=s; cyz*=s; czz*=s;
  float ev[3]; eig3_sym(cxx,cxy,cxz,cyy,cyz,czz,ev);
  float n[3]; evec_smallest(cxx,cxy,cxz,cyy,cyz,czz,ev[0],n);
  float nn = sqrtf(n[0]*n[0]+n[1]*n[1]+n[2]*n[2]) + 1e-12f;
  normals[(size_t)t*3]   = n[0]/nn;
  normals[(size_t)t*3+1] = n[1]/nn;
  normals[(size_t)t*3+2] = n[2]/nn;
}

// kpt -> pts top-16 (includes nearest; matches argsort[:, :, :16])
__global__ void k_knn16(const float* __restrict__ kpt3d, const float* __restrict__ pts,
                        int* __restrict__ idx16) {
  int t = blockIdx.x * blockDim.x + threadIdx.x;
  if (t >= R1_) return;
  int b = t / KPT_;
  float kx = kpt3d[t*3], ky = kpt3d[t*3+1], kz = kpt3d[t*3+2];
  const float* P = pts + (size_t)b * NPTS_ * 3;
  const int K = 16;
  float bd[K]; int bi[K];
  for (int q = 0; q < K; ++q) { bd[q] = 3.4e38f; bi[q] = 0; }
  for (int j = 0; j < NPTS_; ++j) {
    float dx = P[j*3]-kx, dy = P[j*3+1]-ky, dz = P[j*3+2]-kz;
    float d = dx*dx + dy*dy + dz*dz;
    if (d < bd[K-1]) {
      int q = K - 1;
      while (q > 0 && bd[q-1] > d) { bd[q] = bd[q-1]; bi[q] = bi[q-1]; --q; }
      bd[q] = d; bi[q] = j;
    }
  }
  for (int q = 0; q < K; ++q) idx16[t*16 + q] = bi[q];
}

// kpt -> kpt top-8 excluding self; store sqrt distances (reference uses cdist)
__global__ void k_knn8(const float* __restrict__ kpt3d, int* __restrict__ idx8,
                       float* __restrict__ nd8) {
  int t = blockIdx.x * blockDim.x + threadIdx.x;
  if (t >= R1_) return;
  int b = t / KPT_, mm = t - b * KPT_;
  const float* P = kpt3d + (size_t)b * KPT_ * 3;
  float kx = P[mm*3], ky = P[mm*3+1], kz = P[mm*3+2];
  const int K = 8;
  float bd[K]; int bi[K];
  for (int q = 0; q < K; ++q) { bd[q] = 3.4e38f; bi[q] = 0; }
  for (int j = 0; j < KPT_; ++j) {
    if (j == mm) continue;
    float dx = P[j*3]-kx, dy = P[j*3+1]-ky, dz = P[j*3+2]-kz;
    float d = dx*dx + dy*dy + dz*dz;
    if (d < bd[K-1]) {
      int q = K - 1;
      while (q > 0 && bd[q-1] > d) { bd[q] = bd[q-1]; bi[q] = bi[q-1]; --q; }
      bd[q] = d; bi[q] = j;
    }
  }
  for (int q = 0; q < K; ++q) { idx8[t*8+q] = bi[q]; nd8[t*8+q] = sqrtf(bd[q]); }
}

// mean of neighbor normals per keypoint, normalized
__global__ void k_kn(const int* __restrict__ idx16, const float* __restrict__ normals,
                     float* __restrict__ kn) {
  int t = blockIdx.x * blockDim.x + threadIdx.x;
  if (t >= R1_) return;
  int b = t / KPT_;
  float sx = 0, sy = 0, sz = 0;
  for (int q = 0; q < 16; ++q) {
    int id = idx16[t*16+q];
    const float* n = normals + ((size_t)b * NPTS_ + id) * 3;
    sx += n[0]; sy += n[1]; sz += n[2];
  }
  sx *= (1.0f/16); sy *= (1.0f/16); sz *= (1.0f/16);
  float nn = sqrtf(sx*sx + sy*sy + sz*sz) + 1e-12f;
  kn[t*3] = sx/nn; kn[t*3+1] = sy/nn; kn[t*3+2] = sz/nn;
}

// Edge feature assembly: EF bf16 [:,0:128]=center, [:,128:256]=nbr-center; ppf, rel f32
__global__ void k_edge_assemble(const float* __restrict__ kptF, const float* __restrict__ kpt3d,
                                const float* __restrict__ ptsF, const float* __restrict__ pts,
                                const float* __restrict__ normals, const float* __restrict__ kn,
                                const int* __restrict__ idx16,
                                u16* __restrict__ EF, float* __restrict__ ppf,
                                float* __restrict__ relb) {
  int rI = blockIdx.x * blockDim.x + threadIdx.x;
  if (rI >= R16_) return;
  int m = rI >> 4;
  int b = m / KPT_;
  int id = idx16[rI];
  const float* cf = kptF + (size_t)m * C_;
  const float* nf = ptsF + ((size_t)b * NPTS_ + id) * C_;
  u16* e = EF + (size_t)rI * 384;
  for (int c = 0; c < C_; ++c) { float cv = cf[c]; e[c] = f2bf(cv); e[C_ + c] = f2bf(nf[c] - cv); }
  float rx = pts[((size_t)b*NPTS_+id)*3]   - kpt3d[m*3];
  float ry = pts[((size_t)b*NPTS_+id)*3+1] - kpt3d[m*3+1];
  float rz = pts[((size_t)b*NPTS_+id)*3+2] - kpt3d[m*3+2];
  float dn = sqrtf(rx*rx + ry*ry + rz*rz);
  float inv = 1.0f / (dn + 1e-8f);
  float hx = rx*inv, hy = ry*inv, hz = rz*inv;
  float k0 = kn[m*3], k1 = kn[m*3+1], k2 = kn[m*3+2];
  const float* nr = normals + ((size_t)b*NPTS_+id)*3;
  float alpha = fminf(fmaxf(k0*hx + k1*hy + k2*hz, -1.0f), 1.0f);
  float phi   = fminf(fmaxf(nr[0]*hx + nr[1]*hy + nr[2]*hz, -1.0f), 1.0f);
  float theta = fminf(fmaxf(k0*nr[0] + k1*nr[1] + k2*nr[2], -1.0f), 1.0f);
  ppf[(size_t)rI*4]   = dn;   ppf[(size_t)rI*4+1] = alpha;
  ppf[(size_t)rI*4+2] = phi;  ppf[(size_t)rI*4+3] = theta;
  relb[(size_t)rI*3] = rx; relb[(size_t)rI*3+1] = ry; relb[(size_t)rI*3+2] = rz;
}

// Generic small scalar linear: Y[r,o] = X[r,:Kd] . W[o,:Kd] + b[o]
__global__ void k_linear(const float* __restrict__ X, int ldx,
                         const float* __restrict__ W, const float* __restrict__ bias,
                         void* __restrict__ Y, int ldy,
                         int rows, int Kd, int O, int relu, int outBf16) {
  long long t = (long long)blockIdx.x * blockDim.x + threadIdx.x;
  if (t >= (long long)rows * O) return;
  int r = (int)(t / O), o = (int)(t - (long long)r * O);
  float acc = bias[o];
  const float* xr = X + (size_t)r * ldx;
  const float* wr = W + (size_t)o * Kd;
  for (int k = 0; k < Kd; ++k) acc += xr[k] * wr[k];
  if (relu) acc = fmaxf(acc, 0.0f);
  if (outBf16) ((u16*)Y)[(size_t)r * ldy + o] = f2bf(acc);
  else         ((float*)Y)[(size_t)r * ldy + o] = acc;
}

// Batch-norm stats: one block per channel; stats[c]=sum, stats[ch+c]=sumsq
__global__ void k_bn_stats(const float* __restrict__ X, int ldx,
                           float* __restrict__ stats, int rows, int ch) {
  int c = blockIdx.x;
  float s = 0, s2 = 0;
  for (int r = threadIdx.x; r < rows; r += blockDim.x) {
    float v = X[(size_t)r * ldx + c]; s += v; s2 += v * v;
  }
  __shared__ float sh[256], sh2[256];
  sh[threadIdx.x] = s; sh2[threadIdx.x] = s2;
  __syncthreads();
  for (int st = 128; st > 0; st >>= 1) {
    if ((int)threadIdx.x < st) { sh[threadIdx.x] += sh[threadIdx.x+st]; sh2[threadIdx.x] += sh2[threadIdx.x+st]; }
    __syncthreads();
  }
  if (threadIdx.x == 0) { stats[c] = sh[0]; stats[ch + c] = sh2[0]; }
}

__global__ void k_bn_apply(const float* __restrict__ X, int ldx,
                           void* __restrict__ Y, int ldy,
                           const float* __restrict__ stats,
                           const float* __restrict__ g, const float* __restrict__ bb,
                           int rows, int ch, int relu, int outBf16) {
  long long t = (long long)blockIdx.x * blockDim.x + threadIdx.x;
  if (t >= (long long)rows * ch) return;
  int r = (int)(t / ch), c = (int)(t - (long long)r * ch);
  float m = stats[c] / rows;
  float var = stats[ch + c] / rows - m * m;
  float y = (X[(size_t)r*ldx + c] - m) * rsqrtf(var + EPS_BN) * g[c] + bb[c];
  if (relu) y = fmaxf(y, 0.0f);
  if (outBf16) ((u16*)Y)[(size_t)r*ldy + c] = f2bf(y);
  else         ((float*)Y)[(size_t)r*ldy + c] = y;
}

// max over 16 neighbors
__global__ void k_maxpool16(const float* __restrict__ H2, float* __restrict__ local) {
  long long t = (long long)blockIdx.x * blockDim.x + threadIdx.x;
  if (t >= (long long)R1_ * C_) return;
  int m = (int)(t / C_), c = (int)(t - (long long)m * C_);
  float v = -3.4e38f;
  for (int j = 0; j < 16; ++j) v = fmaxf(v, H2[((size_t)m*16 + j) * C_ + c]);
  local[t] = v;
}

// geo features per keypoint (k_graph=8)
__global__ void k_geo(const float* __restrict__ kpt3d, const int* __restrict__ idx8,
                      const float* __restrict__ nd8, float* __restrict__ geo,
                      float* __restrict__ geo_mean) {
  int t = blockIdx.x * blockDim.x + threadIdx.x;
  if (t >= R1_) return;
  int b = t / KPT_;
  const float* P = kpt3d + (size_t)b * KPT_ * 3;
  float nd[8]; int id[8];
  float md = 0;
  for (int j = 0; j < 8; ++j) { nd[j] = nd8[t*8+j]; id[j] = idx8[t*8+j]; md += nd[j]; }
  md *= 0.125f;
  float var = 0;
  for (int j = 0; j < 8; ++j) { float d = nd[j]-md; var += d*d; }
  float sd = sqrtf(var / 7.0f);   // ddof=1
  float cx = 0, cy = 0, cz = 0;
  for (int j = 0; j < 8; ++j) { cx += P[id[j]*3]; cy += P[id[j]*3+1]; cz += P[id[j]*3+2]; }
  cx *= 0.125f; cy *= 0.125f; cz *= 0.125f;
  float rcx = kpt3d[t*3]-cx, rcy = kpt3d[t*3+1]-cy, rcz = kpt3d[t*3+2]-cz;
  float xx=0,xy=0,xz=0,yy=0,yz=0,zz=0;
  for (int j = 0; j < 8; ++j) {
    float dx = P[id[j]*3]-cx, dy = P[id[j]*3+1]-cy, dz = P[id[j]*3+2]-cz;
    xx+=dx*dx; xy+=dx*dy; xz+=dx*dz; yy+=dy*dy; yz+=dy*dz; zz+=dz*dz;
  }
  xx*=0.125f; xy*=0.125f; xz*=0.125f; yy*=0.125f; yz*=0.125f; zz*=0.125f;
  float ev[3]; eig3_sym(xx,xy,xz,yy,yz,zz,ev);
  float aniso = ev[2] / (ev[0] + 1e-8f);
  for (int j = 0; j < 8; ++j) {
    float* gr = geo + ((size_t)t*8 + j) * 7;
    gr[0]=nd[j]; gr[1]=md; gr[2]=sd; gr[3]=rcx; gr[4]=rcy; gr[5]=rcz; gr[6]=aniso;
  }
  float* gm = geo_mean + (size_t)t * 7;
  gm[0]=md; gm[1]=md; gm[2]=sd; gm[3]=rcx; gm[4]=rcy; gm[5]=rcz; gm[6]=aniso;
}

// gather knn keypoint features -> bf16 rows for K/V GEMMs
__global__ void k_gather_kf(const float* __restrict__ kptF, const int* __restrict__ idx8,
                            u16* __restrict__ KF) {
  long long t = (long long)blockIdx.x * blockDim.x + threadIdx.x;
  if (t >= (long long)R8_ * C_) return;
  int row = (int)(t / C_), c = (int)(t - (long long)row * C_);
  int m = row >> 3;
  int b = m / KPT_;
  int id = idx8[row];
  KF[t] = f2bf(kptF[((size_t)b * KPT_ + id) * C_ + c]);
}

// 4-head attention over k=8 neighbors; hd=32; agg written bf16 (feeds oproj GEMM)
__global__ void k_attn(const float* __restrict__ Q, const float* __restrict__ Kb,
                       const float* __restrict__ Vb, const float* __restrict__ GA,
                       u16* __restrict__ agg) {
  int t = blockIdx.x * blockDim.x + threadIdx.x;
  if (t >= R1_ * 4) return;
  int m = t >> 2, h = t & 3;
  const float* q = Q + (size_t)m * C_ + h * 32;
  float s[8], mx = -3.4e38f;
  for (int j = 0; j < 8; ++j) {
    const float* kk = Kb + ((size_t)m*8 + j) * C_ + h * 32;
    float d = 0;
    for (int c = 0; c < 32; ++c) d += q[c] * kk[c];
    s[j] = d * 0.17677669529663689f + GA[((size_t)m*8 + j) * 4 + h];
    mx = fmaxf(mx, s[j]);
  }
  float sum = 0;
  for (int j = 0; j < 8; ++j) { s[j] = expf(s[j] - mx); sum += s[j]; }
  float inv = 1.0f / sum;
  for (int c = 0; c < 32; ++c) {
    float a = 0;
    for (int j = 0; j < 8; ++j) a += s[j] * Vb[((size_t)m*8 + j) * C_ + h*32 + c];
    agg[(size_t)m * C_ + h*32 + c] = f2bf(a * inv);
  }
}

__global__ void k_add(const float* __restrict__ A, const float* __restrict__ Bp,
                      void* __restrict__ Y, long long n, int relu, int outBf16) {
  long long t = (long long)blockIdx.x * blockDim.x + threadIdx.x;
  if (t >= n) return;
  float v = A[t] + Bp[t];
  if (relu) v = fmaxf(v, 0.0f);
  if (outBf16) ((u16*)Y)[t] = f2bf(v);
  else         ((float*)Y)[t] = v;
}

// ---------------------------------------------------------------------------
// Host launch
// ---------------------------------------------------------------------------
static inline int gridFor(long long n, int bs) { return (int)((n + bs - 1) / bs); }

extern "C" void kernel_launch(void* const* d_in, const int* in_sizes, int n_in,
                              void* d_out, int out_size, void* d_ws, size_t ws_size,
                              hipStream_t stream) {
  const float* kptF  = (const float*)d_in[0];   // [8,512,128]
  const float* kpt3d = (const float*)d_in[1];   // [8,512,3]
  const float* ptsF  = (const float*)d_in[2];   // [8,2048,128]
  const float* pts   = (const float*)d_in[3];   // [8,2048,3]
  const float* ppf1_w = (const float*)d_in[4],  *ppf1_b = (const float*)d_in[5];
  const float* ppfbn_g= (const float*)d_in[6],  *ppfbn_b= (const float*)d_in[7];
  const float* ppf2_w = (const float*)d_in[8],  *ppf2_b = (const float*)d_in[9];
  const float* pos_w  = (const float*)d_in[10], *pos_b  = (const float*)d_in[11];
  const float* posbn_g= (const float*)d_in[12], *posbn_b= (const float*)d_in[13];
  const float* e1_w   = (const float*)d_in[14], *e1_b   = (const float*)d_in[15];
  const float* ebn1_g = (const float*)d_in[16], *ebn1_b = (const float*)d_in[17];
  const float* e2_w   = (const float*)d_in[18], *e2_b   = (const float*)d_in[19];
  const float* ebn2_g = (const float*)d_in[20], *ebn2_b = (const float*)d_in[21];
  const float* geo1_w = (const float*)d_in[22], *geo1_b = (const float*)d_in[23];
  const float* geo2_w = (const float*)d_in[24], *geo2_b = (const float*)d_in[25];
  const float* q_w    = (const float*)d_in[26], *q_b    = (const float*)d_in[27];
  const float* k_w    = (const float*)d_in[28], *k_b    = (const float*)d_in[29];
  const float* v_w    = (const float*)d_in[30], *v_b    = (const float*)d_in[31];
  const float* gatt_w = (const float*)d_in[32], *gatt_b = (const float*)d_in[33];
  const float* op_w   = (const float*)d_in[34], *op_b   = (const float*)d_in[35];
  const float* o1_w   = (const float*)d_in[36], *o1_b   = (const float*)d_in[37];
  const float* o2_w   = (const float*)d_in[38], *o2_b   = (const float*)d_in[39];
  float* out = (float*)d_out;                   // [8,512,128]
  (void)in_sizes; (void)n_in; (void)out_size; (void)ws_size;

  // Workspace bump allocator, 256-byte aligned chunks
  char* WS = (char*)d_ws; size_t off = 0;
  auto allocB = [&](size_t bytes) { char* p = WS + off; off += (bytes + 255) & ~(size_t)255; return p; };
  auto allocF = [&](size_t n) { return (float*)allocB(n * 4); };
  auto allocH = [&](size_t n) { return (u16*)allocB(n * 2); };

  float* normals = allocF((size_t)B_*NPTS_*3);
  float* kn      = allocF((size_t)R1_*3);
  int*   idx16   = (int*)allocB(R16_*4);
  int*   idx8    = (int*)allocB(R8_*4);
  float* nd8     = allocF(R8_);
  float* ppf     = allocF((size_t)R16_*4);
  float* relb    = allocF((size_t)R16_*3);
  float* pp1     = allocF((size_t)R16_*32);
  u16*   pp1b    = allocH((size_t)R16_*32);
  float* pe_lin  = allocF((size_t)R16_*64);
  u16*   EF      = allocH((size_t)R16_*384);
  float* H1      = allocF((size_t)R16_*256);
  u16*   H1b     = allocH((size_t)R16_*256);
  float* H2      = allocF((size_t)R16_*128);
  float* local   = allocF((size_t)R1_*C_);
  float* geo     = allocF((size_t)R8_*7);
  float* geom    = allocF((size_t)R1_*7);
  u16*   g1      = allocH((size_t)R1_*64);
  float* geoenc  = allocF((size_t)R1_*C_);
  u16*   KF      = allocH((size_t)R8_*C_);
  u16*   kptFb   = allocH((size_t)R1_*C_);
  float* Qb      = allocF((size_t)R1_*C_);
  float* Kb      = allocF((size_t)R8_*C_);
  float* Vb      = allocF((size_t)R8_*C_);
  float* GA      = allocF((size_t)R8_*4);
  u16*   aggb    = allocH((size_t)R1_*C_);
  float* geoout  = allocF((size_t)R1_*C_);
  u16*   hbuf    = allocH((size_t)R1_*C_);
  u16*   t1      = allocH((size_t)R1_*C_);
  float* t2      = allocF((size_t)R1_*C_);
  float* stats   = allocF(1024);
  // bf16 weight copies (converted once per launch; deterministic)
  u16* w_ppf2 = allocH(64*32);
  u16* w_e1   = allocH((size_t)256*384);
  u16* w_e2   = allocH((size_t)128*256);
  u16* w_geo2 = allocH(128*64);
  u16* w_q    = allocH(128*128);
  u16* w_k    = allocH(128*128);
  u16* w_v    = allocH(128*128);
  u16* w_op   = allocH(128*128);
  u16* w_o1   = allocH(128*128);
  u16* w_o2   = allocH(128*128);

  const int BS = 256;
  auto cvt = [&](const float* X, u16* Y, long long n) {
    hipLaunchKernelGGL(k_cvt_bf16, dim3(gridFor(n, BS)), dim3(BS), 0, stream, X, Y, n);
  };
  auto gemm = [&](const u16* A, const u16* W, const float* bias, void* Cm,
                  int M, int N, int K, int ldC, int relu, int outBf16) {
    int tiles = (M >> 4) * (N >> 6);
    hipLaunchKernelGGL(k_gemm_wmma_bf16, dim3(gridFor(tiles, 8)), dim3(BS), 0, stream,
                       A, W, bias, Cm, M, N, K, ldC, relu, outBf16);
  };

  // 0) weight + feature conversions to bf16
  cvt(ppf2_w, w_ppf2, 64*32);
  cvt(e1_w,   w_e1,   (long long)256*384);
  cvt(e2_w,   w_e2,   (long long)128*256);
  cvt(geo2_w, w_geo2, 128*64);
  cvt(q_w, w_q, 128*128); cvt(k_w, w_k, 128*128); cvt(v_w, w_v, 128*128);
  cvt(op_w, w_op, 128*128); cvt(o1_w, w_o1, 128*128); cvt(o2_w, w_o2, 128*128);
  cvt(kptF, kptFb, (long long)R1_*C_);

  // 1) normals + kNN structures
  hipLaunchKernelGGL(k_normals, dim3(gridFor(B_*NPTS_, BS)), dim3(BS), 0, stream, pts, normals);
  hipLaunchKernelGGL(k_knn16,  dim3(gridFor(R1_, BS)), dim3(BS), 0, stream, kpt3d, pts, idx16);
  hipLaunchKernelGGL(k_knn8,   dim3(gridFor(R1_, BS)), dim3(BS), 0, stream, kpt3d, idx8, nd8);
  hipLaunchKernelGGL(k_kn,     dim3(gridFor(R1_, BS)), dim3(BS), 0, stream, idx16, normals, kn);

  // 2) edge_conv
  hipLaunchKernelGGL(k_edge_assemble, dim3(gridFor(R16_, BS)), dim3(BS), 0, stream,
                     kptF, kpt3d, ptsF, pts, normals, kn, idx16, EF, ppf, relb);
  // ppf MLP: 4->32 (scalar), BN+relu -> bf16, 32->64 (WMMA) into EF[:,256:320]
  hipLaunchKernelGGL(k_linear, dim3(gridFor((long long)R16_*32, BS)), dim3(BS), 0, stream,
                     ppf, 4, ppf1_w, ppf1_b, pp1, 32, R16_, 4, 32, 0, 0);
  hipLaunchKernelGGL(k_bn_stats, dim3(32), dim3(BS), 0, stream, pp1, 32, stats, R16_, 32);
  hipLaunchKernelGGL(k_bn_apply, dim3(gridFor((long long)R16_*32, BS)), dim3(BS), 0, stream,
                     pp1, 32, pp1b, 32, stats, ppfbn_g, ppfbn_b, R16_, 32, 1, 1);
  gemm(pp1b, w_ppf2, ppf2_b, EF + 256, R16_, 64, 32, 384, 0, 1);
  // pos MLP: 3->64 (scalar), BN+relu -> bf16 into EF[:,320:384]
  hipLaunchKernelGGL(k_linear, dim3(gridFor((long long)R16_*64, BS)), dim3(BS), 0, stream,
                     relb, 3, pos_w, pos_b, pe_lin, 64, R16_, 3, 64, 0, 0);
  hipLaunchKernelGGL(k_bn_stats, dim3(64), dim3(BS), 0, stream, pe_lin, 64, stats, R16_, 64);
  hipLaunchKernelGGL(k_bn_apply, dim3(gridFor((long long)R16_*64, BS)), dim3(BS), 0, stream,
                     pe_lin, 64, EF + 320, 384, stats, posbn_g, posbn_b, R16_, 64, 1, 1);
  // edge1: [65536,384] x [256,384]^T (WMMA bf16), BN+relu -> bf16
  gemm(EF, w_e1, e1_b, H1, R16_, 256, 384, 256, 0, 0);
  hipLaunchKernelGGL(k_bn_stats, dim3(256), dim3(BS), 0, stream, H1, 256, stats, R16_, 256);
  hipLaunchKernelGGL(k_bn_apply, dim3(gridFor((long long)R16_*256, BS)), dim3(BS), 0, stream,
                     H1, 256, H1b, 256, stats, ebn1_g, ebn1_b, R16_, 256, 1, 1);
  // edge2: [65536,256] x [128,256]^T (WMMA bf16), BN+relu, maxpool
  gemm(H1b, w_e2, e2_b, H2, R16_, 128, 256, 128, 0, 0);
  hipLaunchKernelGGL(k_bn_stats, dim3(128), dim3(BS), 0, stream, H2, 128, stats, R16_, 128);
  hipLaunchKernelGGL(k_bn_apply, dim3(gridFor((long long)R16_*128, BS)), dim3(BS), 0, stream,
                     H2, 128, H2, 128, stats, ebn2_g, ebn2_b, R16_, 128, 1, 0);
  hipLaunchKernelGGL(k_maxpool16, dim3(gridFor((long long)R1_*C_, BS)), dim3(BS), 0, stream, H2, local);

  // 3) geo_graph
  hipLaunchKernelGGL(k_geo, dim3(gridFor(R1_, BS)), dim3(BS), 0, stream, kpt3d, idx8, nd8, geo, geom);
  hipLaunchKernelGGL(k_linear, dim3(gridFor((long long)R1_*64, BS)), dim3(BS), 0, stream,
                     geom, 7, geo1_w, geo1_b, g1, 64, R1_, 7, 64, 1, 1);
  gemm(g1, w_geo2, geo2_b, geoenc, R1_, 128, 64, 128, 0, 0);
  hipLaunchKernelGGL(k_gather_kf, dim3(gridFor((long long)R8_*C_, BS)), dim3(BS), 0, stream,
                     kptF, idx8, KF);
  gemm(kptFb, w_q, q_b, Qb, R1_, 128, 128, 128, 0, 0);
  gemm(KF,    w_k, k_b, Kb, R8_, 128, 128, 128, 0, 0);
  gemm(KF,    w_v, v_b, Vb, R8_, 128, 128, 128, 0, 0);
  hipLaunchKernelGGL(k_linear, dim3(gridFor((long long)R8_*4, BS)), dim3(BS), 0, stream,
                     geo, 7, gatt_w, gatt_b, GA, 4, R8_, 7, 4, 0, 0);
  hipLaunchKernelGGL(k_attn, dim3(gridFor(R1_*4, BS)), dim3(BS), 0, stream, Qb, Kb, Vb, GA, aggb);
  gemm(aggb, w_op, op_b, geoout, R1_, 128, 128, 128, 0, 0);
  hipLaunchKernelGGL(k_add, dim3(gridFor((long long)R1_*C_, BS)), dim3(BS), 0, stream,
                     geoout, geoenc, geoout, (long long)R1_*C_, 0, 0);

  // 4) fuse + output MLP + residual
  hipLaunchKernelGGL(k_add, dim3(gridFor((long long)R1_*C_, BS)), dim3(BS), 0, stream,
                     local, geoout, hbuf, (long long)R1_*C_, 1, 1);   // relu(local+geo) -> bf16
  gemm(hbuf, w_o1, o1_b, t1, R1_, 128, 128, 128, 1, 1);               // relu(linear) -> bf16
  gemm(t1,   w_o2, o2_b, t2, R1_, 128, 128, 128, 0, 0);
  hipLaunchKernelGGL(k_add, dim3(gridFor((long long)R1_*C_, BS)), dim3(BS), 0, stream,
                     kptF, t2, out, (long long)R1_*C_, 0, 0);         // kpt_feature + h
}
